// DLSMN_ARC_20590073217497
// MI455X (gfx1250) — matrix operand order repaired and checked by reading, not compiled, not run on hardware
//
#include <hip/hip_runtime.h>

// ---------------------------------------------------------------------------
// DLSMN cache-slot-attention forward for MI455X (gfx1250, wave32, WMMA).
// bf16 inputs to v_wmma_f32_16x16x32_bf16 for every GEMM and both attention
// matmuls (flash style); f32 accumulate; f32 VALU epilogues.
// Global->LDS staging uses GLOBAL_LOAD_ASYNC_TO_LDS_B128 (ASYNCcnt) when the
// toolchain exposes the builtin, else plain loads.
// ---------------------------------------------------------------------------

typedef __bf16 bf16;
typedef __attribute__((ext_vector_type(16))) __bf16 v16bf;
typedef __attribute__((ext_vector_type(8)))  __bf16 v8bf;
typedef __attribute__((ext_vector_type(8)))  float  v8f;
typedef int v4i __attribute__((ext_vector_type(4)));

#define B_DIM 8
#define S_DIM 2048
#define DM    1024
#define DC    512
#define KSLOT 256
#define NCACHE 2048           // L*K
#define HHEADS 4
#define HD     128
#define ROW_LO 768            // LAYER_IDX*K
#define ROW_HI 1024

#if defined(__has_builtin)
#if __has_builtin(__builtin_amdgcn_global_load_async_to_lds_b128)
#define HAVE_ASYNC_COPY 1
#endif
#endif

// Copy 16B from global to LDS. Async path bypasses VGPRs (ASYNCcnt-tracked).
static __device__ inline void copy_b128_to_lds(const bf16* g, bf16* l) {
#if HAVE_ASYNC_COPY
  __builtin_amdgcn_global_load_async_to_lds_b128(
      (__attribute__((address_space(1))) v4i*)(g),
      (__attribute__((address_space(3))) v4i*)(l), 0, 0);
#else
  *(v16bf*)l = *(const v16bf*)g;
#endif
}

static __device__ inline void async_copy_wait() {
#if HAVE_ASYNC_COPY
#if __has_builtin(__builtin_amdgcn_s_wait_asynccnt)
  __builtin_amdgcn_s_wait_asynccnt(0);
#else
  asm volatile("s_wait_asynccnt 0" ::: "memory");
#endif
#endif
}

static __device__ inline v8f vzero8() {
  v8f v;
#pragma unroll
  for (int i = 0; i < 8; ++i) v[i] = 0.f;
  return v;
}

static __device__ inline v8f wmma_bf16f32(v16bf a, v16bf b, v8f c) {
  // D = A(16x32 bf16) * B(32x16 bf16) + C(16x16 f32)
  return __builtin_amdgcn_wmma_f32_16x16x32_bf16(false, a, false, b,
                                                 (short)0, c, false, false);
}

// A-fragment (16x32, bf16), contiguous K order (B frags match):
// lane L holds row M = L%16; packed elements 0..15 = K[(L/16)*16 .. +16)
// `row` points at the 32 contiguous K values of this lane's row.
static __device__ inline v16bf load_a_frag(const bf16* __restrict__ row, int lane) {
  const int h = (lane >> 4) * 16;
  v8bf lo = *(const v8bf*)(row + h);
  v8bf hi = *(const v8bf*)(row + h + 8);
  v16bf r;
#pragma unroll
  for (int i = 0; i < 8; ++i) { r[i] = lo[i]; r[i + 8] = hi[i]; }
  return r;
}

// ---------------------------------------------------------------------------
// f32 -> bf16 elementwise convert
// ---------------------------------------------------------------------------
__global__ __launch_bounds__(256) void f32_to_bf16_kernel(
    const float* __restrict__ src, bf16* __restrict__ dst, long n) {
  long i = (long)blockIdx.x * 256 + threadIdx.x;
  const long stride = (long)gridDim.x * 256;
  for (; i < n; i += stride) dst[i] = (bf16)src[i];
}

// ---------------------------------------------------------------------------
// Generic batched bf16 GEMM:  C[bz] = A[bz] (MxK) @ B[bz] (KxN) (+ bias[N])
// Block tile: 256(M) x 64(N); 8 waves as 4(M)x2(N), each wave 64x32 via
// 8 WMMAs per 32-wide K step. Dims must be tile multiples (true here).
// ---------------------------------------------------------------------------
__global__ __launch_bounds__(256) void gemm_bf16_kernel(
    const bf16* __restrict__ A, long sA, int lda,
    const bf16* __restrict__ B, long sB, int ldb,
    float* __restrict__ Cf, bf16* __restrict__ Cb, long sC, int ldc,
    const float* __restrict__ bias, int K, int outBf) {
  __shared__ __align__(32) bf16 At[256][32];   // [m][k]
  __shared__ __align__(32) bf16 Bt[64][32];    // transposed: [n][k]
  const int bz = blockIdx.z;
  A += (long)bz * sA;
  B += (long)bz * sB;
  const int m0 = blockIdx.y * 256;
  const int n0 = blockIdx.x * 64;
  const int tid = threadIdx.x;
  const int lane = tid & 31;
  const int wid = tid >> 5;
  const int wm = (wid & 3) * 64;   // wave M offset in tile
  const int wn = (wid >> 2) * 32;  // wave N offset in tile

  v8f acc[4][2];
#pragma unroll
  for (int i = 0; i < 4; ++i)
#pragma unroll
    for (int j = 0; j < 2; ++j) acc[i][j] = vzero8();

  const int ar = tid >> 1, ac = (tid & 1) * 16;   // A tile: 2 chunks per thread
  for (int k0 = 0; k0 < K; k0 += 32) {
    __syncthreads();
    // A tile: 256x32 bf16, 2x16B per thread, async DMA straight to LDS
    copy_b128_to_lds(A + (long)(m0 + ar) * lda + k0 + ac, &At[ar][ac]);
    copy_b128_to_lds(A + (long)(m0 + ar + 128) * lda + k0 + ac, &At[ar + 128][ac]);
    // B tile, stored transposed [n][k]
    if (tid < 128) {
      const int kr = tid >> 2, c = (tid & 3) * 16;
      const bf16* src = B + (long)(k0 + kr) * ldb + n0 + c;
#pragma unroll
      for (int i = 0; i < 16; ++i) Bt[c + i][kr] = src[i];
      if (k0 + 32 < K)
        __builtin_prefetch(B + (long)(k0 + 32 + kr) * ldb + n0 + c, 0, 1);
    }
    async_copy_wait();
    __syncthreads();

    // B-fragment: lane L holds column N = L%16, K = (L/16)*16 .. +16 contiguous
    v16bf b0 = *(const v16bf*)(&Bt[wn + (lane & 15)][(lane >> 4) * 16]);
    v16bf b1 = *(const v16bf*)(&Bt[wn + 16 + (lane & 15)][(lane >> 4) * 16]);
#pragma unroll
    for (int i = 0; i < 4; ++i) {
      v16bf a = load_a_frag(&At[wm + i * 16 + (lane & 15)][0], lane);
      acc[i][0] = wmma_bf16f32(a, b0, acc[i][0]);
      acc[i][1] = wmma_bf16f32(a, b1, acc[i][1]);
    }
  }

  const long cb = (long)bz * sC;
#pragma unroll
  for (int i = 0; i < 4; ++i)
#pragma unroll
    for (int j = 0; j < 2; ++j) {
      const int nn = n0 + wn + j * 16 + (lane & 15);
      const float bv = bias ? bias[nn] : 0.f;
#pragma unroll
      for (int r = 0; r < 8; ++r) {
        const int mm = m0 + wm + i * 16 + r + 8 * (lane >> 4);
        const float v = acc[i][j][r] + bv;
        if (outBf) Cb[cb + (long)mm * ldc + nn] = (bf16)v;
        else       Cf[cb + (long)mm * ldc + nn] = v;
      }
    }
}

// ---------------------------------------------------------------------------
// Per-token: gate score (dot 1024 + sigmoid) and gumbel-softmax over K=256;
// writes w transposed: wT[b][k][s] (bf16) for the updates GEMM.
// ---------------------------------------------------------------------------
__global__ __launch_bounds__(256) void gate_softmax_kernel(
    const float* __restrict__ y, const float* __restrict__ Wg,
    const float* __restrict__ bg, const float* __restrict__ logits,
    const float* __restrict__ bslot, const float* __restrict__ gamma,
    const float* __restrict__ gu, bf16* __restrict__ wT) {
  __shared__ float red[256];
  const int r = blockIdx.x;      // 0 .. B*S-1
  const int tid = threadIdx.x;
  const float* yr = y + (long)r * DM;
  float s = 0.f;
#pragma unroll
  for (int i = 0; i < 4; ++i) s += yr[tid + i * 256] * Wg[tid + i * 256];
  red[tid] = s;
  __syncthreads();
  for (int off = 128; off > 0; off >>= 1) {
    if (tid < off) red[tid] += red[tid + off];
    __syncthreads();
  }
  const float score = 1.f / (1.f + __expf(-(red[0] + bg[0])));
  __syncthreads();

  const long rk = (long)r * KSLOT + tid;
  const float u = gu[rk];
  const float gn = -__logf(-__logf(u + 1e-8f) + 1e-8f);
  const float z = gamma[0] * (logits[rk] + bslot[tid]) + gn;   // TEMP == 1

  red[tid] = z;
  __syncthreads();
  for (int off = 128; off > 0; off >>= 1) {
    if (tid < off) red[tid] = fmaxf(red[tid], red[tid + off]);
    __syncthreads();
  }
  const float zmax = red[0];
  __syncthreads();
  const float e = __expf(z - zmax);
  red[tid] = e;
  __syncthreads();
  for (int off = 128; off > 0; off >>= 1) {
    if (tid < off) red[tid] += red[tid + off];
    __syncthreads();
  }
  const float p = e / red[0] * score;
  const int b = r >> 11, sidx = r & (S_DIM - 1);
  wT[((long)b * KSLOT + tid) * S_DIM + sidx] = (bf16)p;
}

// mass[b*K+k] = sum_s wT[b][k][s]
__global__ __launch_bounds__(256) void mass_kernel(
    const bf16* __restrict__ wT, float* __restrict__ mass) {
  __shared__ float red[256];
  const int bk = blockIdx.x;
  const int tid = threadIdx.x;
  const bf16* p = wT + (long)bk * S_DIM;
  float s = 0.f;
#pragma unroll
  for (int i = 0; i < 8; ++i) s += (float)p[tid + i * 256];
  red[tid] = s;
  __syncthreads();
  for (int off = 128; off > 0; off >>= 1) {
    if (tid < off) red[tid] += red[tid + off];
    __syncthreads();
  }
  if (tid == 0) mass[bk] = red[0];
}

// cache2 = cache with rows [768,1024) replaced by EMA slot update; also bf16 copy
__global__ __launch_bounds__(256) void cache_update_kernel(
    const float* __restrict__ cache, const float* __restrict__ updates,
    const float* __restrict__ mass, float* __restrict__ cache2,
    bf16* __restrict__ c2b) {
  const long i = (long)blockIdx.x * 256 + threadIdx.x;  // grid covers exactly B*N*DC
  const int d = (int)(i & (DC - 1));
  const long rn = i >> 9;
  const int n = (int)(rn & (NCACHE - 1));
  const int b = (int)(rn >> 11);
  float v = cache[i];
  if (n >= ROW_LO && n < ROW_HI) {
    const int k = n - ROW_LO;
    const float m = mass[b * KSLOT + k];
    const float g = m / (1.f + m);
    const float ns = updates[((long)b * KSLOT + k) * DC + d] / (m + 1e-6f);
    v = (1.f - g) * 0.9f * v + g * ns;
  }
  cache2[i] = v;
  c2b[i] = (bf16)v;
}

// ---------------------------------------------------------------------------
// Flash attention over the 2048-slot cache. Block = 8 waves; each wave owns
// 16 query rows of one (b,h); 32-key tiles staged in LDS; online softmax.
// ---------------------------------------------------------------------------
__global__ __launch_bounds__(256) void attn_kernel(
    const bf16* __restrict__ Q, const bf16* __restrict__ Kc,
    const bf16* __restrict__ Vc, bf16* __restrict__ AO) {
  __shared__ __align__(32) bf16 Kt[32][HD];    // [key][d] (row-major: QK^T B-frag)
  __shared__ __align__(32) bf16 VtT[HD][32];   // [d][key] (transposed: PV B-frag)
  __shared__ __align__(32) bf16 Pb[8][16][32]; // per-wave probabilities
  __shared__ float Ps[8][16][32];              // per-wave raw scores
  __shared__ float rowm[8][16], rowl[8][16], rowf[8][16];

  const int bh = blockIdx.y;
  const int b = bh >> 2, h = bh & 3;
  const int tid = threadIdx.x, lane = tid & 31, w = tid >> 5;
  const int q0 = blockIdx.x * 128 + w * 16;
  const long base = ((long)b * NCACHE) * DC + h * HD;

  v16bf aq[4];
  const bf16* qrow = Q + base + (long)(q0 + (lane & 15)) * DC;
#pragma unroll
  for (int kc = 0; kc < 4; ++kc) aq[kc] = load_a_frag(qrow + kc * 32, lane);

  v8f ao[8];
#pragma unroll
  for (int d = 0; d < 8; ++d) ao[d] = vzero8();
  if (lane < 16) { rowm[w][lane] = -1e30f; rowl[w][lane] = 0.f; }
  const float scale = 0.08838834764831845f;  // 1/sqrt(128)

  const int lr = tid >> 3, lc = (tid & 7) * 16;  // K/V tile loader coords
  for (int t = 0; t < NCACHE / 32; ++t) {
    __syncthreads();
    {
      const long g = base + (long)(t * 32 + lr) * DC + lc;
      copy_b128_to_lds(Kc + g, &Kt[lr][lc]);       // async DMA to LDS
      v16bf vv = *(const v16bf*)(Vc + g);
#pragma unroll
      for (int i = 0; i < 16; ++i) VtT[lc + i][lr] = vv[i];
    }
    async_copy_wait();
    __syncthreads();

    // S = q @ k^T for two 16-key subtiles
#pragma unroll
    for (int ks = 0; ks < 2; ++ks) {
      v8f s = vzero8();
#pragma unroll
      for (int kc = 0; kc < 4; ++kc) {
        v16bf bk = *(const v16bf*)(&Kt[ks * 16 + (lane & 15)][kc * 32 + (lane >> 4) * 16]);
        s = wmma_bf16f32(aq[kc], bk, s);
      }
#pragma unroll
      for (int r = 0; r < 8; ++r)
        Ps[w][r + 8 * (lane >> 4)][ks * 16 + (lane & 15)] = s[r] * scale;
    }

    // Online softmax row stats (wave-internal; LDS ops are in-order per wave)
    if (lane < 16) {
      const float mo = rowm[w][lane];
      float mx = mo;
#pragma unroll
      for (int c = 0; c < 32; ++c) mx = fmaxf(mx, Ps[w][lane][c]);
      const float f = __expf(mo - mx);
      float sum = 0.f;
#pragma unroll
      for (int c = 0; c < 32; ++c) {
        const float e = __expf(Ps[w][lane][c] - mx);
        sum += e;
        Pb[w][lane][c] = (bf16)e;
      }
      rowf[w][lane] = f;
      rowm[w][lane] = mx;
      rowl[w][lane] = rowl[w][lane] * f + sum;
    }

    float fr[8];
#pragma unroll
    for (int r = 0; r < 8; ++r) fr[r] = rowf[w][r + 8 * (lane >> 4)];
#pragma unroll
    for (int d = 0; d < 8; ++d)
#pragma unroll
      for (int r = 0; r < 8; ++r) ao[d][r] *= fr[r];

    // P (16x32) @ V (32x128)
    v16bf ap = load_a_frag(&Pb[w][lane & 15][0], lane);
#pragma unroll
    for (int d = 0; d < 8; ++d) {
      v16bf bv = *(const v16bf*)(&VtT[d * 16 + (lane & 15)][(lane >> 4) * 16]);
      ao[d] = wmma_bf16f32(ap, bv, ao[d]);
    }
  }

  float inv[8];
#pragma unroll
  for (int r = 0; r < 8; ++r) inv[r] = 1.f / rowl[w][r + 8 * (lane >> 4)];
#pragma unroll
  for (int d = 0; d < 8; ++d)
#pragma unroll
    for (int r = 0; r < 8; ++r)
      AO[base + (long)(q0 + r + 8 * (lane >> 4)) * DC + d * 16 + (lane & 15)] =
          (bf16)(ao[d][r] * inv[r]);
}

// out = layernorm(cache2 + oW) * g + b   (population var, eps 1e-5)
__global__ __launch_bounds__(256) void ln_kernel(
    const float* __restrict__ cache2, const float* __restrict__ oW,
    const float* __restrict__ lng, const float* __restrict__ lnb,
    float* __restrict__ out) {
  __shared__ float red[256];
  const int r = blockIdx.x;
  const int tid = threadIdx.x;
  const long base = (long)r * DC;
  const float x0 = cache2[base + tid] + oW[base + tid];
  const float x1 = cache2[base + 256 + tid] + oW[base + 256 + tid];
  red[tid] = x0 + x1;
  __syncthreads();
  for (int off = 128; off > 0; off >>= 1) {
    if (tid < off) red[tid] += red[tid + off];
    __syncthreads();
  }
  const float mu = red[0] * (1.f / DC);
  __syncthreads();
  const float d0 = x0 - mu, d1 = x1 - mu;
  red[tid] = d0 * d0 + d1 * d1;
  __syncthreads();
  for (int off = 128; off > 0; off >>= 1) {
    if (tid < off) red[tid] += red[tid + off];
    __syncthreads();
  }
  const float rs = rsqrtf(red[0] * (1.f / DC) + 1e-5f);
  out[base + tid]       = d0 * rs * lng[tid] + lnb[tid];
  out[base + 256 + tid] = d1 * rs * lng[256 + tid] + lnb[256 + tid];
}

// ---------------------------------------------------------------------------
// Host-side orchestration
// ---------------------------------------------------------------------------
extern "C" void kernel_launch(void* const* d_in, const int* in_sizes, int n_in,
                              void* d_out, int out_size, void* d_ws, size_t ws_size,
                              hipStream_t stream) {
  const float* y       = (const float*)d_in[0];
  const float* cache   = (const float*)d_in[1];
  const float* gu      = (const float*)d_in[2];
  const float* W_gate  = (const float*)d_in[3];
  const float* b_gate  = (const float*)d_in[4];
  const float* W_slot  = (const float*)d_in[5];
  const float* b_slot  = (const float*)d_in[6];
  const float* gamma   = (const float*)d_in[7];
  const float* W_write = (const float*)d_in[8];
  const float* b_write = (const float*)d_in[9];
  const float* Wq      = (const float*)d_in[10];
  const float* bq      = (const float*)d_in[11];
  const float* Wk      = (const float*)d_in[12];
  const float* bk      = (const float*)d_in[13];
  const float* Wv      = (const float*)d_in[14];
  const float* bv      = (const float*)d_in[15];
  const float* Wo      = (const float*)d_in[16];
  const float* bo      = (const float*)d_in[17];
  const float* ln_g    = (const float*)d_in[18];
  const float* ln_b    = (const float*)d_in[19];
  float* out = (float*)d_out;

  char* ws = (char*)d_ws;
  const size_t MB = 1ull << 20;
  // Workspace layout (peak ~160 MB, with reuse of dead regions):
  bf16* yb      = (bf16*)(ws + 0);           // 32MB, dead after logits/write_vals
  bf16* qb      = (bf16*)(ws + 0);           // 16MB (reuse yb)
  bf16* kb      = (bf16*)(ws + 16 * MB);     // 16MB (reuse yb)
  bf16* wslot_b = (bf16*)(ws + 32 * MB);
  bf16* wwrite_b= (bf16*)(ws + 32 * MB + 512 * 1024);
  bf16* wq_b    = (bf16*)(ws + 32 * MB + 1536 * 1024);
  bf16* wk_b    = (bf16*)(ws + 32 * MB + 2048 * 1024);
  bf16* wv_b    = (bf16*)(ws + 32 * MB + 2560 * 1024);
  bf16* wo_b    = (bf16*)(ws + 32 * MB + 3072 * 1024);
  float* logits = (float*)(ws + 36 * MB);    // 16MB, dead after softmax
  bf16* vb      = (bf16*)(ws + 36 * MB);     // 16MB (reuse logits)
  bf16* wvwb    = (bf16*)(ws + 52 * MB);     // 16MB write_vals, dead after updates
  bf16* aob     = (bf16*)(ws + 52 * MB);     // 16MB (reuse wvwb)
  bf16* wT      = (bf16*)(ws + 68 * MB);     // 8MB
  float* updates= (float*)(ws + 76 * MB);    // 4MB
  float* mass   = (float*)(ws + 78 * MB);    // 8KB
  float* cache2 = (float*)(ws + 80 * MB);    // 32MB
  bf16* c2b     = (bf16*)(ws + 112 * MB);    // 16MB
  float* oW     = (float*)(ws + 128 * MB);   // 32MB

  const long MBS = (long)B_DIM * S_DIM;      // 16384
  const long MBN = (long)B_DIM * NCACHE;     // 16384

  // 1) bf16 conversions
  f32_to_bf16_kernel<<<8192, 256, 0, stream>>>(y, yb, MBS * DM);
  f32_to_bf16_kernel<<<256, 256, 0, stream>>>(W_slot, wslot_b, (long)DM * KSLOT);
  f32_to_bf16_kernel<<<512, 256, 0, stream>>>(W_write, wwrite_b, (long)DM * DC);
  f32_to_bf16_kernel<<<256, 256, 0, stream>>>(Wq, wq_b, (long)DC * DC);
  f32_to_bf16_kernel<<<256, 256, 0, stream>>>(Wk, wk_b, (long)DC * DC);
  f32_to_bf16_kernel<<<256, 256, 0, stream>>>(Wv, wv_b, (long)DC * DC);
  f32_to_bf16_kernel<<<256, 256, 0, stream>>>(Wo, wo_b, (long)DC * DC);

  // 2) logits = y @ W_slot      (f32 out; gamma/b_slot applied in softmax)
  gemm_bf16_kernel<<<dim3(KSLOT / 64, MBS / 256, 1), 256, 0, stream>>>(
      yb, 0, DM, wslot_b, 0, KSLOT, logits, nullptr, 0, KSLOT, nullptr, DM, 0);
  // 3) write_vals = y @ W_write + b_write   (bf16 out)
  gemm_bf16_kernel<<<dim3(DC / 64, MBS / 256, 1), 256, 0, stream>>>(
      yb, 0, DM, wwrite_b, 0, DC, nullptr, wvwb, 0, DC, b_write, DM, 1);

  // 4) gate + gumbel softmax -> wT (bf16, [b][k][s]); 5) mass
  gate_softmax_kernel<<<(int)MBS, 256, 0, stream>>>(
      y, W_gate, b_gate, logits, b_slot, gamma, gu, wT);
  mass_kernel<<<B_DIM * KSLOT, 256, 0, stream>>>(wT, mass);

  // 6) updates[b] = wT[b] (256x2048) @ write_vals[b] (2048x512)
  gemm_bf16_kernel<<<dim3(DC / 64, KSLOT / 256, B_DIM), 256, 0, stream>>>(
      wT, (long)KSLOT * S_DIM, S_DIM,
      wvwb, (long)S_DIM * DC, DC,
      updates, nullptr, (long)KSLOT * DC, DC, nullptr, S_DIM, 0);

  // 7) cache slot EMA update -> cache2 (f32) + bf16 copy
  cache_update_kernel<<<(int)((long)B_DIM * NCACHE * DC / 256), 256, 0, stream>>>(
      cache, updates, mass, cache2, c2b);

  // 8) Q/K/V projections (bf16 out)
  gemm_bf16_kernel<<<dim3(DC / 64, MBN / 256, 1), 256, 0, stream>>>(
      c2b, 0, DC, wq_b, 0, DC, nullptr, qb, 0, DC, bq, DC, 1);
  gemm_bf16_kernel<<<dim3(DC / 64, MBN / 256, 1), 256, 0, stream>>>(
      c2b, 0, DC, wk_b, 0, DC, nullptr, kb, 0, DC, bk, DC, 1);
  gemm_bf16_kernel<<<dim3(DC / 64, MBN / 256, 1), 256, 0, stream>>>(
      c2b, 0, DC, wv_b, 0, DC, nullptr, vb, 0, DC, bv, DC, 1);

  // 9) flash attention -> ao (bf16)
  attn_kernel<<<dim3(NCACHE / 128, B_DIM * HHEADS), 256, 0, stream>>>(
      qb, kb, vb, aob);

  // 10) output projection (f32 out)
  gemm_bf16_kernel<<<dim3(DC / 64, MBN / 256, 1), 256, 0, stream>>>(
      aob, 0, DC, wo_b, 0, DC, oW, nullptr, 0, DC, bo, DC, 0);

  // 11) residual + layernorm -> d_out
  ln_kernel<<<(int)MBN, 256, 0, stream>>>(cache2, oW, ln_g, ln_b, out);
}
